// FlashbackPlusPlus_16965120819731
// MI455X (gfx1250) — compile-verified, matrix operand order checked by reading, not run on hardware
//
#include <hip/hip_runtime.h>
#include <hip/hip_bf16.h>
#include <math.h>

#define S_LEN 256
#define U_LEN 64
#define H_DIM 256
#define V_DIM 10000
#define H3    768
#define K2H   512
#define NROWS (S_LEN * U_LEN)   // 16384

typedef __bf16 bf16_t;
typedef __attribute__((ext_vector_type(16))) __bf16 v16bf;
typedef __attribute__((ext_vector_type(8)))  __bf16 v8bf;
typedef __attribute__((ext_vector_type(8)))  float  v8f;

union Frag16 {
    v16bf  v;
    v8bf   h[2];
    bf16_t e[16];
};

// ---------------------------------------------------------------------------
// f32 -> bf16 elementwise conversion
// ---------------------------------------------------------------------------
__global__ void k_f32_to_bf16(const float* __restrict__ src,
                              bf16_t* __restrict__ dst, int n) {
    int i = blockIdx.x * blockDim.x + threadIdx.x;
    if (i < n) dst[i] = (bf16_t)src[i];
}

// ---------------------------------------------------------------------------
// gi = emb(x) @ W_ih^T + b_ih    A gathered by x; one wave per 16x16 tile
// A: [16384 x 256] bf16 (gathered rows of enc_W), B[k][n] = W_ih[n][k]
// ---------------------------------------------------------------------------
__global__ __launch_bounds__(32)
void k_gi_gemm(const int* __restrict__ x,
               const bf16_t* __restrict__ encW,
               const bf16_t* __restrict__ Wih,
               const float* __restrict__ b_ih,
               float* __restrict__ gi) {
    const int lane  = threadIdx.x;
    const int rbase = blockIdx.x * 16;
    const int nbase = blockIdx.y * 16;
    const int m  = lane & 15;
    const int kh = (lane >> 4) << 3;

    const bf16_t* arow = encW + (size_t)x[rbase + m] * H_DIM;
    const bf16_t* brow = Wih  + (size_t)(nbase + m) * H_DIM;

    v8f acc = {};
#pragma unroll
    for (int k0 = 0; k0 < H_DIM; k0 += 32) {
        Frag16 a, b;
        a.h[0] = *(const v8bf*)(arow + k0 + kh);
        a.h[1] = *(const v8bf*)(arow + k0 + kh + 16);
        b.h[0] = *(const v8bf*)(brow + k0 + kh);
        b.h[1] = *(const v8bf*)(brow + k0 + kh + 16);
        acc = __builtin_amdgcn_wmma_f32_16x16x32_bf16(false, a.v, false, b.v,
                                                      (short)0, acc, false, false);
    }
    const int n  = lane & 15;
    const int mo = (lane >> 4) << 3;
#pragma unroll
    for (int j = 0; j < 8; j++) {
        int r = rbase + j + mo;
        gi[(size_t)r * H3 + nbase + n] = acc[j] + b_ih[nbase + n];
    }
}

// ---------------------------------------------------------------------------
// Persistent GRU scan: ONE workgroup (32 waves = 1024 threads) runs all 256
// steps. Hidden state lives in LDS twice:
//   - f32 ping-pong (2 x 64KB) for the gate blend + h_last
//   - bf16 mirror ping-pong (2 x 32KB), written at production time, so each
//     WMMA A-fragment is two direct ds_load_b128s (no f32->bf16 cvt chain on
//     the sequential critical path).
// Each wave owns 2 of the 64 (16x16) tiles of [U x H]; per step per tile it
// runs 3 WMMA chains (r/z/n gates, K=256) against W_hh.  While computing step
// s, all 1024 threads prefetch step s+1's gi slice (global_prefetch_b8).
// Hidden states are emitted bf16-TRANSPOSED as out_bT[u][h][s] for the
// attention GEMM's B operand.
// ---------------------------------------------------------------------------
__global__ __launch_bounds__(1024)
void k_gru_scan(const float* __restrict__ h0,
                const float* __restrict__ gi,
                const bf16_t* __restrict__ Whh,
                const float* __restrict__ b_hh,
                bf16_t* __restrict__ out_bT,
                float* __restrict__ hlast) {
    extern __shared__ char smem_raw[];
    float*  hs0 = (float*)smem_raw;                                   // 64 KB
    float*  hs1 = (float*)(smem_raw + 65536);                         // 64 KB
    bf16_t* hb0 = (bf16_t*)(smem_raw + 131072);                       // 32 KB
    bf16_t* hb1 = (bf16_t*)(smem_raw + 163840);                       // 32 KB

    const int tid  = threadIdx.x;
    const int wave = tid >> 5;
    const int lane = tid & 31;
    const int m  = lane & 15;
    const int kh = (lane >> 4) << 3;
    const int mo = (lane >> 4) << 3;

    for (int i = tid; i < U_LEN * H_DIM; i += 1024) {
        float v = h0[i];
        hs0[i] = v;
        hb0[i] = (bf16_t)v;
    }
    __syncthreads();

    for (int s = 0; s < S_LEN; s++) {
        const float*  hprev  = (s & 1) ? hs1 : hs0;
        float*        hnext  = (s & 1) ? hs0 : hs1;
        const bf16_t* hbprev = (s & 1) ? hb1 : hb0;
        bf16_t*       hbnext = (s & 1) ? hb0 : hb1;
        const float*  gi_s   = gi + (size_t)s * U_LEN * H3;

        // Prefetch next step's gi slice (U*3H*4 = 196608 B = 1024 threads*192B)
        if (s + 1 < S_LEN) {
            const char* nxt = (const char*)(gi + (size_t)(s + 1) * U_LEN * H3);
            __builtin_prefetch(nxt + (size_t)tid * 192, 0, 0);
        }

#pragma unroll
        for (int tt = 0; tt < 2; tt++) {
            const int tile  = wave * 2 + tt;
            const int ubase = (tile & 3) * 16;
            const int cbase = (tile >> 2) * 16;

            const bf16_t* hrowb = hbprev + (size_t)(ubase + m) * H_DIM;
            const bf16_t* br_r  = Whh + (size_t)(0 * H_DIM + cbase + m) * H_DIM;
            const bf16_t* br_z  = Whh + (size_t)(1 * H_DIM + cbase + m) * H_DIM;
            const bf16_t* br_n  = Whh + (size_t)(2 * H_DIM + cbase + m) * H_DIM;

            v8f ar = {}, az = {}, an = {};
#pragma unroll
            for (int k0 = 0; k0 < H_DIM; k0 += 32) {
                Frag16 a;
                a.h[0] = *(const v8bf*)(hrowb + k0 + kh);
                a.h[1] = *(const v8bf*)(hrowb + k0 + kh + 16);
                Frag16 b;
                b.h[0] = *(const v8bf*)(br_r + k0 + kh);
                b.h[1] = *(const v8bf*)(br_r + k0 + kh + 16);
                ar = __builtin_amdgcn_wmma_f32_16x16x32_bf16(
                        false, a.v, false, b.v, (short)0, ar, false, false);
                b.h[0] = *(const v8bf*)(br_z + k0 + kh);
                b.h[1] = *(const v8bf*)(br_z + k0 + kh + 16);
                az = __builtin_amdgcn_wmma_f32_16x16x32_bf16(
                        false, a.v, false, b.v, (short)0, az, false, false);
                b.h[0] = *(const v8bf*)(br_n + k0 + kh);
                b.h[1] = *(const v8bf*)(br_n + k0 + kh + 16);
                an = __builtin_amdgcn_wmma_f32_16x16x32_bf16(
                        false, a.v, false, b.v, (short)0, an, false, false);
            }

            const int c = cbase + (lane & 15);
#pragma unroll
            for (int j = 0; j < 8; j++) {
                int u = ubase + j + mo;
                float ghr = ar[j] + b_hh[c];
                float ghz = az[j] + b_hh[H_DIM + c];
                float ghn = an[j] + b_hh[2 * H_DIM + c];
                const float* gir = gi_s + (size_t)u * H3;
                float r  = 1.f / (1.f + expf(-(gir[c] + ghr)));
                float z  = 1.f / (1.f + expf(-(gir[H_DIM + c] + ghz)));
                float nn = tanhf(gir[2 * H_DIM + c] + r * ghn);
                float hp = hprev[(size_t)u * H_DIM + c];
                float hv = (1.f - z) * nn + z * hp;
                hnext[(size_t)u * H_DIM + c]  = hv;
                hbnext[(size_t)u * H_DIM + c] = (bf16_t)hv;
                out_bT[((size_t)u * H_DIM + c) * S_LEN + s] = (bf16_t)hv;
            }
        }
        __syncthreads();
    }
    // 256 steps: final write (s=255, odd) landed in hs0
    for (int i = tid; i < U_LEN * H_DIM; i += 1024) hlast[i] = hs0[i];
}

// ---------------------------------------------------------------------------
// w[u,i,j] = f_t(t_i - t_j) * f_s(||s_i - s_j||) + 1e-10 (causal), bf16,
// plus row sums. grid(U), block(S)
// ---------------------------------------------------------------------------
__global__ void k_weights(const float* __restrict__ t,
                          const float* __restrict__ s2,
                          bf16_t* __restrict__ wbuf,
                          float* __restrict__ sumw) {
    const int u = blockIdx.x;
    const int i = threadIdx.x;
    const float ti  = t[i * U_LEN + u];
    const float six = s2[((size_t)i * U_LEN + u) * 2 + 0];
    const float siy = s2[((size_t)i * U_LEN + u) * 2 + 1];
    bf16_t* wrow = wbuf + ((size_t)u * S_LEN + i) * S_LEN;
    float sum = 0.f;
    for (int j = 0; j < S_LEN; j++) {
        float w = 0.f;
        if (j <= i) {
            float dt = ti - t[j * U_LEN + u];
            float dx = six - s2[((size_t)j * U_LEN + u) * 2 + 0];
            float dy = siy - s2[((size_t)j * U_LEN + u) * 2 + 1];
            float ds = sqrtf(dx * dx + dy * dy);
            float ft = (cosf(6.283185307f * dt) + 1.f) * 0.5f * expf(-0.1f * dt);
            float fs = expf(-100.f * ds);
            w = ft * fs + 1e-10f;
            sum += w;
        }
        wrow[j] = (bf16_t)w;
    }
    sumw[i * U_LEN + u] = sum;
}

// ---------------------------------------------------------------------------
// out_w = (w @ out) / sum_w  -> bf16 into out_pu[:, 0:H]
// grid(U, S/16, H/16), block 32.
// A row i of w (contiguous in j); B column h from out_bT[u][h][:] (contiguous
// in j as well thanks to the transposed store in the GRU kernel).
// ---------------------------------------------------------------------------
__global__ __launch_bounds__(32)
void k_attn_gemm(const bf16_t* __restrict__ wbuf,
                 const bf16_t* __restrict__ out_bT,
                 const float* __restrict__ sumw,
                 bf16_t* __restrict__ outpu) {
    const int lane  = threadIdx.x;
    const int u     = blockIdx.x;
    const int ibase = blockIdx.y * 16;
    const int hbase = blockIdx.z * 16;
    const int m  = lane & 15;
    const int kh = (lane >> 4) << 3;

    const bf16_t* arow = wbuf   + ((size_t)u * S_LEN + ibase + m) * S_LEN;
    const bf16_t* bcol = out_bT + ((size_t)u * H_DIM + hbase + m) * S_LEN;

    v8f acc = {};
#pragma unroll
    for (int k0 = 0; k0 < S_LEN; k0 += 32) {
        Frag16 a, b;
        a.h[0] = *(const v8bf*)(arow + k0 + kh);
        a.h[1] = *(const v8bf*)(arow + k0 + kh + 16);
        b.h[0] = *(const v8bf*)(bcol + k0 + kh);
        b.h[1] = *(const v8bf*)(bcol + k0 + kh + 16);
        acc = __builtin_amdgcn_wmma_f32_16x16x32_bf16(false, a.v, false, b.v,
                                                      (short)0, acc, false, false);
    }
    const int n  = lane & 15;
    const int mo = (lane >> 4) << 3;
#pragma unroll
    for (int j = 0; j < 8; j++) {
        int i = ibase + j + mo;
        float inv = 1.f / sumw[i * U_LEN + u];
        outpu[((size_t)i * U_LEN + u) * K2H + hbase + n] = (bf16_t)(acc[j] * inv);
    }
}

// ---------------------------------------------------------------------------
// out_pu[:, H:2H] = user_W[active_user[u]]  (bf16)   grid(S*U), block(H)
// ---------------------------------------------------------------------------
__global__ void k_fill_user(const int* __restrict__ active_user,
                            const float* __restrict__ userW,
                            bf16_t* __restrict__ outpu) {
    const int r = blockIdx.x;        // r = i*U + u
    const int c = threadIdx.x;
    const int u = r & (U_LEN - 1);
    outpu[(size_t)r * K2H + H_DIM + c] =
        (bf16_t)userW[(size_t)active_user[u] * H_DIM + c];
}

// ---------------------------------------------------------------------------
// y = out_pu @ fc_W^T + fc_b     M=16384 N=10000 K=512
// grid(M/64, N/16), block 32; each wave does 4 M-tiles reusing the B fragment
// ---------------------------------------------------------------------------
__global__ __launch_bounds__(32)
void k_fc(const bf16_t* __restrict__ outpu,
          const bf16_t* __restrict__ fcW,
          const float* __restrict__ fcb,
          float* __restrict__ y) {
    const int lane  = threadIdx.x;
    const int rbase = blockIdx.x * 64;
    const int nbase = blockIdx.y * 16;
    const int m  = lane & 15;
    const int kh = (lane >> 4) << 3;

    const bf16_t* brow = fcW + (size_t)(nbase + m) * K2H;
    const bf16_t* a0 = outpu + (size_t)(rbase + m) * K2H;
    const bf16_t* a1 = a0 + (size_t)16 * K2H;
    const bf16_t* a2 = a0 + (size_t)32 * K2H;
    const bf16_t* a3 = a0 + (size_t)48 * K2H;

    v8f c0 = {}, c1 = {}, c2 = {}, c3 = {};
#pragma unroll
    for (int k0 = 0; k0 < K2H; k0 += 32) {
        Frag16 b;
        b.h[0] = *(const v8bf*)(brow + k0 + kh);
        b.h[1] = *(const v8bf*)(brow + k0 + kh + 16);
        Frag16 a;
        a.h[0] = *(const v8bf*)(a0 + k0 + kh);
        a.h[1] = *(const v8bf*)(a0 + k0 + kh + 16);
        c0 = __builtin_amdgcn_wmma_f32_16x16x32_bf16(false, a.v, false, b.v,
                                                     (short)0, c0, false, false);
        a.h[0] = *(const v8bf*)(a1 + k0 + kh);
        a.h[1] = *(const v8bf*)(a1 + k0 + kh + 16);
        c1 = __builtin_amdgcn_wmma_f32_16x16x32_bf16(false, a.v, false, b.v,
                                                     (short)0, c1, false, false);
        a.h[0] = *(const v8bf*)(a2 + k0 + kh);
        a.h[1] = *(const v8bf*)(a2 + k0 + kh + 16);
        c2 = __builtin_amdgcn_wmma_f32_16x16x32_bf16(false, a.v, false, b.v,
                                                     (short)0, c2, false, false);
        a.h[0] = *(const v8bf*)(a3 + k0 + kh);
        a.h[1] = *(const v8bf*)(a3 + k0 + kh + 16);
        c3 = __builtin_amdgcn_wmma_f32_16x16x32_bf16(false, a.v, false, b.v,
                                                     (short)0, c3, false, false);
    }
    const int n  = lane & 15;
    const int mo = (lane >> 4) << 3;
    const float bias = fcb[nbase + n];
#pragma unroll
    for (int j = 0; j < 8; j++) {
        int mm = j + mo;
        y[(size_t)(rbase + mm)      * V_DIM + nbase + n] = c0[j] + bias;
        y[(size_t)(rbase + 16 + mm) * V_DIM + nbase + n] = c1[j] + bias;
        y[(size_t)(rbase + 32 + mm) * V_DIM + nbase + n] = c2[j] + bias;
        y[(size_t)(rbase + 48 + mm) * V_DIM + nbase + n] = c3[j] + bias;
    }
}

// ---------------------------------------------------------------------------
extern "C" void kernel_launch(void* const* d_in, const int* in_sizes, int n_in,
                              void* d_out, int out_size, void* d_ws, size_t ws_size,
                              hipStream_t stream) {
    // inputs (setup_inputs order)
    const int*   x       = (const int*)  d_in[0];
    const float* t       = (const float*)d_in[1];
    const float* s2      = (const float*)d_in[2];
    // d_in[3] y_t, d_in[4] y_s unused
    const float* h0      = (const float*)d_in[5];
    const int*   act_usr = (const int*)  d_in[6];
    const float* enc_W   = (const float*)d_in[7];
    const float* user_W  = (const float*)d_in[8];
    const float* W_ih    = (const float*)d_in[9];
    const float* W_hh    = (const float*)d_in[10];
    const float* b_ih    = (const float*)d_in[11];
    const float* b_hh    = (const float*)d_in[12];
    const float* fc_W    = (const float*)d_in[13];
    const float* fc_b    = (const float*)d_in[14];

    // workspace carve-up
    char*  ws  = (char*)d_ws;
    size_t off = 0;
    auto carve = [&](size_t bytes) -> char* {
        char* p = ws + off;
        off = (off + bytes + 255) & ~(size_t)255;
        return p;
    };
    bf16_t* encW_b = (bf16_t*)carve((size_t)V_DIM * H_DIM * 2);
    bf16_t* Wih_b  = (bf16_t*)carve((size_t)H3 * H_DIM * 2);
    bf16_t* Whh_b  = (bf16_t*)carve((size_t)H3 * H_DIM * 2);
    bf16_t* fcW_b  = (bf16_t*)carve((size_t)V_DIM * K2H * 2);
    float*  gi     = (float*) carve((size_t)NROWS * H3 * 4);
    bf16_t* out_bT = (bf16_t*)carve((size_t)U_LEN * H_DIM * S_LEN * 2);
    bf16_t* wbuf   = (bf16_t*)carve((size_t)U_LEN * S_LEN * S_LEN * 2);
    float*  sumw   = (float*) carve((size_t)NROWS * 4);
    bf16_t* outpu  = (bf16_t*)carve((size_t)NROWS * K2H * 2);
    (void)ws_size; (void)n_in; (void)in_sizes; (void)out_size;

    float* y     = (float*)d_out;
    float* hlast = y + (size_t)NROWS * V_DIM;

    // 1) weight conversions f32 -> bf16
    {
        int n;
        n = V_DIM * H_DIM;
        k_f32_to_bf16<<<(n + 255) / 256, 256, 0, stream>>>(enc_W, encW_b, n);
        n = H3 * H_DIM;
        k_f32_to_bf16<<<(n + 255) / 256, 256, 0, stream>>>(W_ih, Wih_b, n);
        k_f32_to_bf16<<<(n + 255) / 256, 256, 0, stream>>>(W_hh, Whh_b, n);
        n = V_DIM * K2H;
        k_f32_to_bf16<<<(n + 255) / 256, 256, 0, stream>>>(fc_W, fcW_b, n);
    }

    // 2) gi = emb(x) @ W_ih^T + b_ih
    k_gi_gemm<<<dim3(NROWS / 16, H3 / 16), 32, 0, stream>>>(x, encW_b, Wih_b,
                                                            b_ih, gi);

    // 3) persistent GRU scan: one workgroup, h (f32 + bf16 mirror) in 192KB
    //    LDS, 256 steps, gi prefetch one step ahead; writes out_bT and h_last
    k_gru_scan<<<1, 1024, 196608, stream>>>(h0, gi, Whh_b, b_hh, out_bT, hlast);

    // 4) attention weights + row sums
    k_weights<<<U_LEN, S_LEN, 0, stream>>>(t, s2, wbuf, sumw);

    // 5) out_w = (w @ out)/sum_w -> out_pu[:, :H]
    k_attn_gemm<<<dim3(U_LEN, S_LEN / 16, H_DIM / 16), 32, 0, stream>>>(
        wbuf, out_bT, sumw, outpu);

    // 6) out_pu[:, H:2H] = user embedding
    k_fill_user<<<NROWS, H_DIM, 0, stream>>>(act_usr, user_W, outpu);

    // 7) y = out_pu @ fc_W^T + fc_b   (writes directly into d_out)
    k_fc<<<dim3(NROWS / 64, V_DIM / 16), 32, 0, stream>>>(outpu, fcW_b, fc_b, y);
}
